// MultiHeadAttention_38285338476673
// MI455X (gfx1250) — matrix operand order repaired and checked by reading, not compiled, not run on hardware
//
#include <hip/hip_runtime.h>
#include <hip/hip_bf16.h>
#include <math.h>

// ---------------------------------------------------------------------------
// MHA forward for MI455X (gfx1250): bf16 WMMA (16x16x32, f32 acc) everywhere,
// flash-attention fusion so the 192x1024x1024 score matrix never hits HBM.
// Each wave owns TWO 16-row M-strips so every B/K/V fragment load feeds two
// WMMAs (loads-per-WMMA ~1.25 in the GEMM mainloop).
// ---------------------------------------------------------------------------

typedef __bf16 bf16_t;
typedef __attribute__((ext_vector_type(16))) __bf16 v16bf;
typedef __attribute__((ext_vector_type(8)))  __bf16 v8bf;
typedef __attribute__((ext_vector_type(8)))  float  v8f;

#define WMMA_BF16(a, b, c) \
  __builtin_amdgcn_wmma_f32_16x16x32_bf16(false, (a), false, (b), (short)0, (c), false, false)

static constexpr int BATCH = 16;
static constexpr int SEQ   = 1024;
static constexpr int EMB   = 768;
static constexpr int HEADS = 12;
static constexpr int HD    = 64;
static constexpr float ATT_SCALE = 0.03608439182435161f; // 768^-0.5 (module scales by emb_dim)

__device__ __forceinline__ v8f vzero8() {
  v8f z = {0.f, 0.f, 0.f, 0.f, 0.f, 0.f, 0.f, 0.f};
  return z;
}

// A-fragment (16x32 bf16) load, contiguous K order:
//  lanes 0-15 : row = lane,    K = k0+0..15
//  lanes 16-31: row = lane-16, K = k0+16..31
__device__ __forceinline__ v16bf load_afrag(const bf16_t* __restrict__ rowptr, int k0, int half) {
  v8bf lo = *(const v8bf*)(rowptr + k0 + half * 16);
  v8bf hi = *(const v8bf*)(rowptr + k0 + half * 16 + 8);
  v16bf a;
#pragma unroll
  for (int i = 0; i < 8; ++i) { a[i] = lo[i]; a[i + 8] = hi[i]; }
  return a;
}

// ---------------------------------------------------------------------------
// fp32 -> bf16 bulk conversion, 8 elements/thread (b128 in, b128 out)
// ---------------------------------------------------------------------------
__global__ __launch_bounds__(256) void cvt_f32_bf16_kernel(const float* __restrict__ src,
                                                           bf16_t* __restrict__ dst, int n8) {
  const int i = blockIdx.x * blockDim.x + threadIdx.x;
  if (i < n8) {
    const float4 f0 = ((const float4*)src)[2 * i];
    const float4 f1 = ((const float4*)src)[2 * i + 1];
    v8bf o;
    o[0] = (bf16_t)f0.x; o[1] = (bf16_t)f0.y; o[2] = (bf16_t)f0.z; o[3] = (bf16_t)f0.w;
    o[4] = (bf16_t)f1.x; o[5] = (bf16_t)f1.y; o[6] = (bf16_t)f1.z; o[7] = (bf16_t)f1.w;
    ((v8bf*)dst)[i] = o;
  }
}

// ---------------------------------------------------------------------------
// QKV GEMM: C[16384,2304] = X[16384,768] @ Wqkv^T + b_qkv
// Block = 256 threads = 8 waves; tile 256(M) x 128(N); wave = two 16-row strips.
// Epilogue scatters into Q,K: [B,H,N,64] bf16 and V transposed: [B,H,64,N].
// ---------------------------------------------------------------------------
__global__ __launch_bounds__(256)
void qkv_gemm_kernel(const bf16_t* __restrict__ X, const bf16_t* __restrict__ W,
                     const float* __restrict__ bias,
                     bf16_t* __restrict__ Q, bf16_t* __restrict__ Kc, bf16_t* __restrict__ Vt) {
  const int lane = threadIdx.x & 31;
  const int wave = threadIdx.x >> 5;
  const int half = lane >> 4;
  const int l16  = lane & 15;
  const int row0 = blockIdx.y * 256 + wave * 32;
  const int col0 = blockIdx.x * 128;

  const bf16_t* Arow0 = X + (size_t)(row0 + l16) * EMB;
  const bf16_t* Arow1 = X + (size_t)(row0 + 16 + l16) * EMB;

  v8f acc0[8], acc1[8];
#pragma unroll
  for (int nt = 0; nt < 8; ++nt) { acc0[nt] = vzero8(); acc1[nt] = vzero8(); }

  for (int k0 = 0; k0 < EMB; k0 += 32) {
    const v16bf a0 = load_afrag(Arow0, k0, half);
    const v16bf a1 = load_afrag(Arow1, k0, half);
#pragma unroll
    for (int nt = 0; nt < 8; ++nt) {
      // B-frag: lane = output col (a row of W), K contiguous, split by half-wave
      const bf16_t* Brow = W + (size_t)(col0 + nt * 16 + l16) * EMB + k0 + half * 16;
      const v16bf b = *(const v16bf*)Brow;
      acc0[nt] = WMMA_BF16(a0, b, acc0[nt]);
      acc1[nt] = WMMA_BF16(a1, b, acc1[nt]);
    }
  }

#pragma unroll
  for (int s = 0; s < 2; ++s) {
    const v8f* acc = s ? acc1 : acc0;
    const int rowbase = row0 + s * 16;
#pragma unroll
    for (int nt = 0; nt < 8; ++nt) {
      const int n = col0 + nt * 16 + l16;     // 0..2303
      const int which = n / EMB;              // 0=q 1=k 2=v
      const int d = n - which * EMB;
      const int h = d >> 6;
      const int hd_i = d & 63;
      const float bs = bias[n];
#pragma unroll
      for (int r = 0; r < 8; ++r) {
        const int m = rowbase + half * 8 + r; // global row = b*SEQ + nseq
        const int bidx = m >> 10;
        const int nseq = m & 1023;
        const bf16_t val = (bf16_t)(acc[nt][r] + bs);
        const size_t bh = (size_t)(bidx * HEADS + h);
        if (which == 0)      Q [(bh * SEQ + nseq) * HD + hd_i] = val;
        else if (which == 1) Kc[(bh * SEQ + nseq) * HD + hd_i] = val;
        else                 Vt[(bh * HD + hd_i) * SEQ + nseq] = val;   // transposed
      }
    }
  }
}

// ---------------------------------------------------------------------------
// Flash attention: grid (SEQ/256, B*H). 8 waves x two 16-row query strips.
// Per 32-key step: 8 WMMA for Q@K^T, online softmax (16-lane shfl reductions),
// P converted C-layout -> A-layout via per-wave LDS tile, 8 WMMA for P@V.
// K/V fragments are loaded once and reused by both query strips.
// ---------------------------------------------------------------------------
__global__ __launch_bounds__(256)
void attn_kernel(const bf16_t* __restrict__ Q, const bf16_t* __restrict__ Kc,
                 const bf16_t* __restrict__ Vt, bf16_t* __restrict__ AO) {
  __shared__ __align__(32) bf16_t Plds[8][32][32];   // per-wave P tiles (2 strips x 16 x 32)

  const int lane = threadIdx.x & 31;
  const int wave = threadIdx.x >> 5;
  const int half = lane >> 4;
  const int l16  = lane & 15;
  const int bh   = blockIdx.y;
  const int b    = bh / HEADS;
  const int h    = bh - b * HEADS;
  const int q0   = blockIdx.x * 256 + wave * 32;

  const bf16_t* Qb = Q  + (size_t)bh * SEQ * HD;
  const bf16_t* Kb = Kc + (size_t)bh * SEQ * HD;
  const bf16_t* Vb = Vt + (size_t)bh * HD * SEQ;

  // Q fragments for this wave's 2 strips (hd = 0..31 and 32..63)
  v16bf qa[2][2];
#pragma unroll
  for (int s = 0; s < 2; ++s) {
    const bf16_t* Qrow = Qb + (size_t)(q0 + s * 16 + l16) * HD;
    qa[s][0] = load_afrag(Qrow, 0, half);
    qa[s][1] = load_afrag(Qrow, 32, half);
  }

  v8f o[2][4];
  float mrow[2][8], lrow[2][8];
#pragma unroll
  for (int s = 0; s < 2; ++s) {
#pragma unroll
    for (int nt = 0; nt < 4; ++nt) o[s][nt] = vzero8();
#pragma unroll
    for (int r = 0; r < 8; ++r) { mrow[s][r] = -1e30f; lrow[s][r] = 0.f; }
  }

  for (int kb = 0; kb < SEQ; kb += 32) {
    // S = Q @ K^T : B-frag lane = key column, kd contiguous (K rows contiguous)
    const bf16_t* K0 = Kb + (size_t)(kb + l16) * HD + half * 16;
    const bf16_t* K1 = Kb + (size_t)(kb + 16 + l16) * HD + half * 16;
    const v16bf kf00 = *(const v16bf*)(K0);        // keys kb..kb+15,  kd 0..31
    const v16bf kf01 = *(const v16bf*)(K0 + 32);   // keys kb..kb+15,  kd 32..63
    const v16bf kf10 = *(const v16bf*)(K1);        // keys kb+16..31,  kd 0..31
    const v16bf kf11 = *(const v16bf*)(K1 + 32);   // keys kb+16..31,  kd 32..63

#pragma unroll
    for (int s = 0; s < 2; ++s) {
      v8f s0 = vzero8(), s1 = vzero8();
      s0 = WMMA_BF16(qa[s][0], kf00, s0);
      s0 = WMMA_BF16(qa[s][1], kf01, s0);
      s1 = WMMA_BF16(qa[s][0], kf10, s1);
      s1 = WMMA_BF16(qa[s][1], kf11, s1);

      // Online softmax per row (row values live across the 16-lane half-group)
#pragma unroll
      for (int r = 0; r < 8; ++r) {
        float x0 = s0[r] * ATT_SCALE;
        float x1 = s1[r] * ATT_SCALE;
        float rmax = fmaxf(x0, x1);
        rmax = fmaxf(rmax, __shfl_xor(rmax, 1));
        rmax = fmaxf(rmax, __shfl_xor(rmax, 2));
        rmax = fmaxf(rmax, __shfl_xor(rmax, 4));
        rmax = fmaxf(rmax, __shfl_xor(rmax, 8));
        const float mN = fmaxf(mrow[s][r], rmax);
        const float p0 = __expf(x0 - mN);
        const float p1 = __expf(x1 - mN);
        float ps = p0 + p1;
        ps += __shfl_xor(ps, 1);
        ps += __shfl_xor(ps, 2);
        ps += __shfl_xor(ps, 4);
        ps += __shfl_xor(ps, 8);
        const float corr = __expf(mrow[s][r] - mN);
        lrow[s][r] = lrow[s][r] * corr + ps;
        mrow[s][r] = mN;
#pragma unroll
        for (int nt = 0; nt < 4; ++nt) o[s][nt][r] *= corr;
        // C-layout -> row-major P tile in LDS (this wave's private region)
        Plds[wave][s * 16 + half * 8 + r][l16]      = (bf16_t)p0;
        Plds[wave][s * 16 + half * 8 + r][16 + l16] = (bf16_t)p1;
      }
    }
    __builtin_amdgcn_wave_barrier();

    // Reload P tiles as A-fragments (16 rows x 32 keys per strip)
    v16bf pa[2];
#pragma unroll
    for (int s = 0; s < 2; ++s) {
      const bf16_t* Pr = &Plds[wave][s * 16 + l16][0];
      v8bf plo = *(const v8bf*)(Pr + half * 16);
      v8bf phi = *(const v8bf*)(Pr + half * 16 + 8);
#pragma unroll
      for (int i = 0; i < 8; ++i) { pa[s][i] = plo[i]; pa[s][i + 8] = phi[i]; }
    }
    __builtin_amdgcn_wave_barrier();

    // O += P @ V  (V stored transposed [hd][seq] -> contiguous B-frags, reused
    // by both strips)
#pragma unroll
    for (int nt = 0; nt < 4; ++nt) {
      const bf16_t* Vrow = Vb + (size_t)(nt * 16 + l16) * SEQ + kb + half * 16;
      const v16bf vf = *(const v16bf*)Vrow;
      o[0][nt] = WMMA_BF16(pa[0], vf, o[0][nt]);
      o[1][nt] = WMMA_BF16(pa[1], vf, o[1][nt]);
    }
  }

  // Normalize and store attention output as [B, N, H*hd] bf16 (rows feed proj)
#pragma unroll
  for (int s = 0; s < 2; ++s) {
#pragma unroll
    for (int nt = 0; nt < 4; ++nt) {
#pragma unroll
      for (int r = 0; r < 8; ++r) {
        const int qn = q0 + s * 16 + half * 8 + r;
        const float v = o[s][nt][r] / lrow[s][r];
        AO[((size_t)(b * SEQ + qn)) * EMB + h * HD + nt * 16 + l16] = (bf16_t)v;
      }
    }
  }
}

// ---------------------------------------------------------------------------
// Output projection: Out[16384,768] = AO[16384,768] @ Wproj^T + b_proj (fp32)
// ---------------------------------------------------------------------------
__global__ __launch_bounds__(256)
void proj_gemm_kernel(const bf16_t* __restrict__ A, const bf16_t* __restrict__ W,
                      const float* __restrict__ bias, float* __restrict__ Out) {
  const int lane = threadIdx.x & 31;
  const int wave = threadIdx.x >> 5;
  const int half = lane >> 4;
  const int l16  = lane & 15;
  const int row0 = blockIdx.y * 256 + wave * 32;
  const int col0 = blockIdx.x * 128;

  const bf16_t* Arow0 = A + (size_t)(row0 + l16) * EMB;
  const bf16_t* Arow1 = A + (size_t)(row0 + 16 + l16) * EMB;

  v8f acc0[8], acc1[8];
#pragma unroll
  for (int nt = 0; nt < 8; ++nt) { acc0[nt] = vzero8(); acc1[nt] = vzero8(); }

  for (int k0 = 0; k0 < EMB; k0 += 32) {
    const v16bf a0 = load_afrag(Arow0, k0, half);
    const v16bf a1 = load_afrag(Arow1, k0, half);
#pragma unroll
    for (int nt = 0; nt < 8; ++nt) {
      const bf16_t* Brow = W + (size_t)(col0 + nt * 16 + l16) * EMB + k0 + half * 16;
      const v16bf b = *(const v16bf*)Brow;
      acc0[nt] = WMMA_BF16(a0, b, acc0[nt]);
      acc1[nt] = WMMA_BF16(a1, b, acc1[nt]);
    }
  }

#pragma unroll
  for (int s = 0; s < 2; ++s) {
    const v8f* acc = s ? acc1 : acc0;
    const int rowbase = row0 + s * 16;
#pragma unroll
    for (int nt = 0; nt < 8; ++nt) {
      const int n = col0 + nt * 16 + l16;
      const float bs = bias[n];
#pragma unroll
      for (int r = 0; r < 8; ++r) {
        const int m = rowbase + half * 8 + r;
        Out[(size_t)m * EMB + n] = acc[nt][r] + bs;
      }
    }
  }
}

// ---------------------------------------------------------------------------
// Host launcher
// ---------------------------------------------------------------------------
extern "C" void kernel_launch(void* const* d_in, const int* in_sizes, int n_in,
                              void* d_out, int out_size, void* d_ws, size_t ws_size,
                              hipStream_t stream) {
  (void)in_sizes; (void)n_in; (void)out_size; (void)ws_size;
  const float* x      = (const float*)d_in[0];
  const float* w_qkv  = (const float*)d_in[1];
  const float* b_qkv  = (const float*)d_in[2];
  const float* w_proj = (const float*)d_in[3];
  const float* b_proj = (const float*)d_in[4];

  char* ws = (char*)d_ws;
  size_t off = 0;
  auto carve = [&](size_t bytes) -> void* {
    void* p = ws + off;
    off += (bytes + 255) & ~(size_t)255;
    return p;
  };
  const size_t NX = (size_t)BATCH * SEQ * EMB;        // 12,582,912
  bf16_t* Xb     = (bf16_t*)carve(NX * 2);
  bf16_t* Wqkvb  = (bf16_t*)carve((size_t)3 * EMB * EMB * 2);
  bf16_t* Wprojb = (bf16_t*)carve((size_t)EMB * EMB * 2);
  bf16_t* Qb     = (bf16_t*)carve(NX * 2);            // [B,H,N,64]
  bf16_t* Kb     = (bf16_t*)carve(NX * 2);            // [B,H,N,64]
  bf16_t* Vtb    = (bf16_t*)carve(NX * 2);            // [B,H,64,N] (transposed)
  bf16_t* AOb    = (bf16_t*)carve(NX * 2);            // [B,N,768]

  // 1) fp32 -> bf16 conversions (8 elems/thread)
  {
    int n8 = (int)(NX / 8);
    cvt_f32_bf16_kernel<<<dim3((n8 + 255) / 256), 256, 0, stream>>>(x, Xb, n8);
    n8 = 3 * EMB * EMB / 8;
    cvt_f32_bf16_kernel<<<dim3((n8 + 255) / 256), 256, 0, stream>>>(w_qkv, Wqkvb, n8);
    n8 = EMB * EMB / 8;
    cvt_f32_bf16_kernel<<<dim3((n8 + 255) / 256), 256, 0, stream>>>(w_proj, Wprojb, n8);
  }

  // 2) QKV projection: tiles (2304/128, 16384/256)
  qkv_gemm_kernel<<<dim3(18, 64), 256, 0, stream>>>(Xb, Wqkvb, b_qkv, Qb, Kb, Vtb);

  // 3) Flash attention: (1024/256 query blocks, B*H heads)
  attn_kernel<<<dim3(4, BATCH * HEADS), 256, 0, stream>>>(Qb, Kb, Vtb, AOb);

  // 4) Output projection: tiles (768/128, 16384/256)
  proj_gemm_kernel<<<dim3(6, 64), 256, 0, stream>>>(AOb, Wprojb, b_proj, (float*)d_out);
}
